// FutureSceneEncoder_69209103008093
// MI455X (gfx1250) — matrix operand order repaired and checked by reading, not compiled, not run on hardware
//
#include <hip/hip_runtime.h>
#include <math.h>

typedef __attribute__((ext_vector_type(16))) _Float16 v16h;
typedef __attribute__((ext_vector_type(8)))  float    v8f;

#define NSCENE 128
#define NAG    64
#define INDIM  32
#define EMB    64
#define NLAYER 4
#define ENCD   32
#define HP     65            // padded row stride (bank-conflict free)

// dynamic-LDS float offsets
#define OFF_H     0                   // 64*HP f32 node features
#define OFF_AGGR  (64*HP)             // 64*HP f32 message sums
#define OFF_DIST  (2*64*HP)           // 64*HP f32 pairwise distances
#define OFF_AC    (3*64*HP)           // 2*64*HP f32: Apre|Cpre, reused as Wu1[128x64]
#define OFF_W     (5*64*HP)           // 9216 f32 staging (Wm1+bm1+bm2 | Wu2+bu1+bu2+mid)
#define SW_FLOATS 9216
#define SMEM_FLOATS (5*64*HP + SW_FLOATS)            // 30016 floats
#define OFF_WM2P  (SMEM_FLOATS*4)                    // byte offset of packed f16 Wm2
#define SMEM_BYTES (OFF_WM2P + 4096*2)               // +8KB f16 -> 128256 B total

// Branch-free tanh from hardware transcendentals (v_exp_f32 + v_rcp_f32).
// tanh(x) = 1 - 2/(exp(2x)+1); exact at saturation via inf arithmetic.
__device__ __forceinline__ float tanh_fast(float x) {
  float e = __builtin_amdgcn_exp2f(x * 2.8853900817779268f);  // exp(2x)
  return 1.0f - 2.0f * __builtin_amdgcn_rcpf(e + 1.0f);
}

__global__ __launch_bounds__(256) void scene_enc_kernel(
    const float* __restrict__ pos,  const float* __restrict__ x_enc,
    const float* __restrict__ W_in, const float* __restrict__ b_in,
    const float* __restrict__ Wm1,  const float* __restrict__ bm1,
    const float* __restrict__ Wm2,  const float* __restrict__ bm2,
    const float* __restrict__ Wu1,  const float* __restrict__ bu1,
    const float* __restrict__ Wu2,  const float* __restrict__ bu2,
    const float* __restrict__ Wf1,  const float* __restrict__ bf1,
    const float* __restrict__ Wf2,  const float* __restrict__ bf2,
    const int*   __restrict__ T,    float* __restrict__ out)
{
  extern __shared__ __align__(32) char smemraw[];
  float* smem  = (float*)smemraw;
  float* sH    = smem + OFF_H;
  float* sAggr = smem + OFF_AGGR;
  float* sDist = smem + OFF_DIST;
  float* sAC   = smem + OFF_AC;
  float* sW    = smem + OFF_W;
  _Float16* sWm2p = (_Float16*)(smemraw + OFF_WM2P);

  const int b    = blockIdx.x;
  const int tid  = threadIdx.x;
  const int lane = tid & 31;
  const int wv   = tid >> 5;     // wave 0..7
  const int node = tid & 63;
  const int kg   = tid >> 6;     // 0..3 : which 16-wide output column group
  const int half = lane >> 4;

  // ---------- pairwise distances (once per scene) ----------
  for (int idx = tid; idx < NAG*NAG; idx += 256) {
    int i = idx >> 6, j = idx & 63;
    float dx = pos[b*NAG*2 + i*2 + 0] - pos[b*NAG*2 + j*2 + 0];
    float dy = pos[b*NAG*2 + i*2 + 1] - pos[b*NAG*2 + j*2 + 1];
    sDist[i*HP + j] = sqrtf(dx*dx + dy*dy);
  }
  // stage W_in (32x64) + b_in
  for (int idx = tid; idx < (INDIM*EMB)/4; idx += 256)
    ((float4*)sW)[idx] = ((const float4*)W_in)[idx];
  if (tid < EMB) sW[INDIM*EMB + tid] = b_in[tid];
  __syncthreads();

  // ---------- lin_in: sH = x_enc[b] @ W_in + b_in ----------
  {
    float acc[16];
#pragma unroll
    for (int c = 0; c < 16; ++c) acc[c] = 0.f;
    const float* xr = x_enc + (b*NAG + node) * INDIM;
    for (int m = 0; m < INDIM; ++m) {
      float xv = xr[m];
      const float* w = sW + m*EMB + kg*16;
#pragma unroll
      for (int c = 0; c < 16; ++c) acc[c] += xv * w[c];
    }
#pragma unroll
    for (int c = 0; c < 16; ++c)
      sH[node*HP + kg*16 + c] = acc[c] + sW[INDIM*EMB + kg*16 + c];
  }
  const int tcls = T[b*NAG + node];

  // =================== message-passing layers ===================
  for (int l = 0; l < NLAYER; ++l) {
    __syncthreads();
    // stage Wm1[l] (135x64) -> sW[0..8640), bm1 -> 8640, bm2 -> 8704
    {
      const float* wm1 = Wm1 + l * (2*EMB + 7) * EMB;
      for (int idx = tid; idx < ((2*EMB+7)*EMB)/4; idx += 256)
        ((float4*)sW)[idx] = ((const float4*)wm1)[idx];
      if (tid < EMB) {
        sW[8640 + tid] = bm1[l*EMB + tid];
        sW[8704 + tid] = bm2[l*EMB + tid];
      }
      // pack Wm2[l] into B-fragment layout in this kernel's contiguous k order (32x16 f16 tiles, wave32):
      // tile(kt,nt): lane L -> N = nt*16+(L&15); element e -> K = kt*32 + e + 16*(L>>4)
      const float* wm2 = Wm2 + l*EMB*EMB;
      {
        int tile = tid >> 5, lam = tid & 31;
        int kt = tile >> 2, nt = tile & 3;
        int n  = nt*16 + (lam & 15);
        int kb = kt*32 + 16*(lam >> 4);
#pragma unroll
        for (int e = 0; e < 16; ++e)
          sWm2p[tid*16 + e] = (_Float16)wm2[(kb + e)*EMB + n];
      }
      for (int idx = tid; idx < 64*HP; idx += 256) sAggr[idx] = 0.f;
    }
    __syncthreads();

    // per-node edge-MLP terms (f32 VALU; only 3% of FLOPs, keeps precision)
    // Apre_i = h_i@Wm1[0:64]   + Wm1[131+T_i] + bm1
    // Cpre_j = h_j@Wm1[64:128] + Wm1[128+T_j]
    {
      float accA[16], accC[16];
#pragma unroll
      for (int c = 0; c < 16; ++c) { accA[c] = 0.f; accC[c] = 0.f; }
      const float* hrow = sH + node*HP;
      for (int m = 0; m < EMB; ++m) {
        float hv = hrow[m];
        const float* w1 = sW + m*EMB + kg*16;
        const float* w2 = sW + (EMB + m)*EMB + kg*16;
#pragma unroll
        for (int c = 0; c < 16; ++c) { accA[c] += hv*w1[c]; accC[c] += hv*w2[c]; }
      }
      const float* rA = sW + (131 + tcls)*EMB + kg*16;
      const float* rC = sW + (128 + tcls)*EMB + kg*16;
      const float* b1 = sW + 8640 + kg*16;
#pragma unroll
      for (int c = 0; c < 16; ++c) {
        sAC[node*HP + kg*16 + c]         = accA[c] + rA[c] + b1[c];
        sAC[64*HP + node*HP + kg*16 + c] = accC[c] + rC[c];
      }
    }
    __syncthreads();

    if (l + 1 < NLAYER)   // warm L2 for next layer's big weight block
      __builtin_prefetch(Wm1 + (l+1)*(2*EMB+7)*EMB + lane*64, 0, 0);

    // ---------- edge phase: per dst node a 64x64x64 GEMM on WMMA ----------
    {
      const float* sA = sAC;
      const float* sC = sAC + 64*HP;
      const float* wd = sW + 134*EMB;             // dist row of Wm1
      const v16h*  bP = (const v16h*)sWm2p;
#pragma unroll 1
      for (int q = 0; q < 8; ++q) {
        int i = wv*8 + q;                          // destination node, uniform per wave
#pragma unroll 1
        for (int ms = 0; ms < 4; ++ms) {           // M-strip: source rows ms*16..+15
          int   j   = ms*16 + (lane & 15);
          float dij = sDist[i*HP + j];
          v8f accs[4] = {v8f{}, v8f{}, v8f{}, v8f{}};
#pragma unroll
          for (int kt = 0; kt < 2; ++kt) {
            // A fragment: t1[j,k] = tanh(Apre_i[k] + Cpre_j[k] + dij*Wd[k])
            v16h a;
            int kb = kt*32 + 16*half;
#pragma unroll
            for (int e = 0; e < 16; ++e) {
              int k = kb + e;
              float v = sA[i*HP + k] + sC[j*HP + k] + dij * wd[k];
              a[e] = (_Float16)tanh_fast(v);
            }
#pragma unroll
            for (int nt = 0; nt < 4; ++nt)
              accs[nt] = __builtin_amdgcn_wmma_f32_16x16x32_f16(
                  false, a, false, bP[(kt*4 + nt)*32 + lane],
                  (short)0, accs[nt], false, false);
          }
          // msg = tanh(acc + bm2); aggregate over source rows into LDS
#pragma unroll
          for (int nt = 0; nt < 4; ++nt) {
            int n = nt*16 + (lane & 15);
            float bias = sW[8704 + n];
            float s = 0.f;
#pragma unroll
            for (int r = 0; r < 8; ++r) s += tanh_fast(accs[nt][r] + bias);
            atomicAdd(&sAggr[i*HP + n], s);        // ds_add_f32
          }
        }
      }
    }
    __syncthreads();

    // ---------- node update MLP (f32 VALU; tiny) ----------
    {
      const float* wu1 = Wu1 + l*2*EMB*EMB;
      for (int idx = tid; idx < (2*EMB*EMB)/4; idx += 256)
        ((float4*)sAC)[idx] = ((const float4*)wu1)[idx];
      const float* wu2 = Wu2 + l*EMB*EMB;
      for (int idx = tid; idx < (EMB*EMB)/4; idx += 256)
        ((float4*)sW)[idx] = ((const float4*)wu2)[idx];
      if (tid < EMB) {
        sW[4096 + tid] = bu1[l*EMB + tid];
        sW[4160 + tid] = bu2[l*EMB + tid];
      }
    }
    __syncthreads();
    {
      float acc[16];
#pragma unroll
      for (int c = 0; c < 16; ++c) acc[c] = 0.f;
      for (int m = 0; m < EMB; ++m) {
        float hv = sH[node*HP + m];
        const float* w = sAC + m*EMB + kg*16;
#pragma unroll
        for (int c = 0; c < 16; ++c) acc[c] += hv * w[c];
      }
      for (int m = 0; m < EMB; ++m) {
        float av = sAggr[node*HP + m];
        const float* w = sAC + (EMB + m)*EMB + kg*16;
#pragma unroll
        for (int c = 0; c < 16; ++c) acc[c] += av * w[c];
      }
      float* mid = sW + 4224;
#pragma unroll
      for (int c = 0; c < 16; ++c)
        mid[node*HP + kg*16 + c] = tanh_fast(acc[c] + sW[4096 + kg*16 + c]);
    }
    __syncthreads();
    {
      float acc[16];
#pragma unroll
      for (int c = 0; c < 16; ++c) acc[c] = 0.f;
      const float* mid = sW + 4224;
      for (int m = 0; m < EMB; ++m) {
        float mv = mid[node*HP + m];
        const float* w = sW + m*EMB + kg*16;
#pragma unroll
        for (int c = 0; c < 16; ++c) acc[c] += mv * w[c];
      }
#pragma unroll
      for (int c = 0; c < 16; ++c)
        sH[node*HP + kg*16 + c] += tanh_fast(acc[c] + sW[4160 + kg*16 + c]);
    }
  } // layers

  // ---------- mean pool + encoder FC ----------
  __syncthreads();
  if (tid < EMB) {
    float s = 0.f;
    for (int i = 0; i < NAG; ++i) s += sH[i*HP + tid];
    sAC[tid] = s * (1.f/64.f);
  }
  __syncthreads();
  if (tid < ENCD) {
    float s = 0.f;
    for (int m = 0; m < EMB; ++m) s += sAC[m] * Wf1[m*ENCD + tid];
    sAC[128 + tid] = tanh_fast(s + bf1[tid]);
  }
  __syncthreads();
  if (tid < ENCD) {
    float s = 0.f;
    for (int m = 0; m < ENCD; ++m) s += sAC[128 + m] * Wf2[m*ENCD + tid];
    out[b*ENCD + tid] = s + bf2[tid];
  }
}

extern "C" void kernel_launch(void* const* d_in, const int* in_sizes, int n_in,
                              void* d_out, int out_size, void* d_ws, size_t ws_size,
                              hipStream_t stream) {
  const float* pos   = (const float*)d_in[0];
  const float* x_enc = (const float*)d_in[1];
  // d_in[2] pos_emb, d_in[3] numAgents_emb: unused by the reference
  const float* W_in  = (const float*)d_in[4];
  const float* b_in  = (const float*)d_in[5];
  const float* Wm1   = (const float*)d_in[6];
  const float* bm1   = (const float*)d_in[7];
  const float* Wm2   = (const float*)d_in[8];
  const float* bm2   = (const float*)d_in[9];
  const float* Wu1   = (const float*)d_in[10];
  const float* bu1   = (const float*)d_in[11];
  const float* Wu2   = (const float*)d_in[12];
  const float* bu2   = (const float*)d_in[13];
  const float* Wf1   = (const float*)d_in[14];
  const float* bf1   = (const float*)d_in[15];
  const float* Wf2   = (const float*)d_in[16];
  const float* bf2   = (const float*)d_in[17];
  const int*   T     = (const int*)d_in[18];
  float* out = (float*)d_out;

  scene_enc_kernel<<<NSCENE, 256, SMEM_BYTES, stream>>>(
      pos, x_enc, W_in, b_in, Wm1, bm1, Wm2, bm2,
      Wu1, bu1, Wu2, bu2, Wf1, bf1, Wf2, bf2, T, out);
}
